// TransformerBlock_55190329754232
// MI455X (gfx1250) — compile-verified, matrix operand order checked by reading, not run on hardware
//
#include <hip/hip_runtime.h>
#include <hip/hip_bf16.h>
#include <math.h>

// ---------------------------------------------------------------------------
// CDNA5 / gfx1250 transformer block.  All heavy math through
// v_wmma_f32_16x16x32_bf16 (fp32 accumulate).  Wave32 fragment layouts per
// the CDNA5 ISA tables (CDNA5 ISA):
//   A 16x32 bf16 : lane half selects K base (0 / 8); VGPR j<4 -> K=kb+2j,
//                  VGPR 4+j -> K=16+kb+2j, row M = lane&15.
//   B 32x16 bf16 : mirror of A with N = lane&15 (B staged transposed in LDS
//                  so pairs are contiguous ds_load_b32).
//   C/D 16x16 f32: VGPR r -> M = r + (lane>=16 ? 8 : 0), N = lane&15.
// Tile staging uses GLOBAL_LOAD_ASYNC_TO_LDS (ASYNCcnt) and the Tensor Data
// Mover (TENSORcnt) where the toolchain exposes them; falls back to
// b128-load + ds_store otherwise.
// ---------------------------------------------------------------------------

#if defined(__has_builtin)
# if __has_builtin(__builtin_amdgcn_global_load_async_to_lds_b128)
#  define HAVE_ASYNC_LDS 1
# endif
# if __has_builtin(__builtin_amdgcn_tensor_load_to_lds) && \
     __has_builtin(__builtin_amdgcn_s_wait_tensorcnt)
#  define HAVE_TDM 1
# endif
#endif
#ifndef HAVE_ASYNC_LDS
# define HAVE_ASYNC_LDS 0
#endif
#ifndef HAVE_TDM
# define HAVE_TDM 0
#endif

typedef __attribute__((ext_vector_type(16))) __bf16 v16bf;
typedef __attribute__((ext_vector_type(8)))  float  v8f;
typedef __attribute__((ext_vector_type(4)))  int    v4i_t;

union FragBF { v16bf v; unsigned u[8]; unsigned short s[16]; };

__device__ __forceinline__ unsigned short f2bf(float f) {
    unsigned u = __float_as_uint(f);
    unsigned r = u + 0x7FFFu + ((u >> 16) & 1u);   // round-to-nearest-even
    return (unsigned short)(r >> 16);
}

// 16-byte global -> LDS copy: async (ASYNCcnt) when available.
// Builtin signature (from toolchain diagnostic): param0 = v4i in global
// (__device__/AS1) address space, param1 = LDS side, then imm offset, cpol.
__device__ __forceinline__ void stage16(const unsigned short* g, unsigned short* l) {
#if HAVE_ASYNC_LDS
    __builtin_amdgcn_global_load_async_to_lds_b128(
        (__attribute__((address_space(1))) v4i_t*)g,
        (__attribute__((address_space(3))) v4i_t*)l, 0, 0);
#else
    const uint4 v = *(const uint4*)g;
    *(unsigned*)(l + 0) = v.x; *(unsigned*)(l + 2) = v.y;
    *(unsigned*)(l + 4) = v.z; *(unsigned*)(l + 6) = v.w;
#endif
}

__device__ __forceinline__ void stage_fence() {
#if HAVE_ASYNC_LDS
# if __has_builtin(__builtin_amdgcn_s_wait_asynccnt)
    __builtin_amdgcn_s_wait_asynccnt(0);
# else
    asm volatile("s_wait_asynccnt 0x0" ::: "memory");
# endif
#endif
}

// ---------------------------------------------------------------------------
// fp32 -> bf16 cast with 2-D zero padding.
// ---------------------------------------------------------------------------
__global__ void cast_pad_bf16(const float* __restrict__ src,
                              unsigned short* __restrict__ dst,
                              int srows, int scols, int drows, int dcols) {
    int idx = blockIdx.x * blockDim.x + threadIdx.x;
    int total = drows * dcols;
    if (idx >= total) return;
    int r = idx / dcols, c = idx - r * dcols;
    float v = (r < srows && c < scols) ? src[(size_t)r * scols + c] : 0.f;
    dst[idx] = f2bf(v);
}

// ---------------------------------------------------------------------------
// Tiled bf16 GEMM, C = A(MxK) * B(KxN), 128x128 block, 8 waves, 32-k steps.
// mode 0: store bf16; mode 1: +bias, leaky_relu(0.2), bf16; mode 2: +bias, f32
// ---------------------------------------------------------------------------
#define GLP 34   // padded LDS pitch in bf16 units (68B -> conflict-free)

__global__ __launch_bounds__(256)
void gemm_bf16_wmma(const unsigned short* __restrict__ A,
                    const unsigned short* __restrict__ B,
                    const float* __restrict__ bias, int biasN,
                    float* __restrict__ Cf, unsigned short* __restrict__ Cb,
                    int M, int N, int K, int mode)
{
    __shared__ __align__(16) unsigned short sA[128 * GLP];
    __shared__ __align__(16) unsigned short sBt[128 * GLP];

    const int tid  = threadIdx.x;
    const int lane = tid & 31;
    const int wave = tid >> 5;
    const int half = lane >> 4;
    const int lm   = lane & 15;
    const int kb   = half * 8;
    const int mbase = blockIdx.y * 128;
    const int nbase = blockIdx.x * 128;
    const int wm = (wave & 3) * 32;
    const int wn = (wave >> 2) * 64;

    v8f acc[2][4];
    for (int i = 0; i < 2; ++i) for (int j = 0; j < 4; ++j) acc[i][j] = {};

    const int ksteps = K >> 5;
    for (int kt = 0; kt < ksteps; ++kt) {
        // --- stage A 128x32 (row major, contiguous 16B chunks -> async) ---
        for (int i = 0; i < 2; ++i) {
            int c    = tid + 256 * i;          // 0..511 chunks of 8 bf16
            int row  = c >> 2;
            int coff = (c & 3) * 8;
            stage16(A + (size_t)(mbase + row) * K + kt * 32 + coff,
                    &sA[row * GLP + coff]);
        }
        // --- stage B 32x128 transposed -> sBt[n][k] (scatter: manual) ---
        for (int i = 0; i < 2; ++i) {
            int c  = tid + 256 * i;
            int kr = c >> 4;                   // 0..31
            int nc = (c & 15) * 8;
            const uint4 vb = *(const uint4*)(B + (size_t)(kt * 32 + kr) * N + nbase + nc);
            unsigned short t[8];
            *(unsigned*)&t[0] = vb.x; *(unsigned*)&t[2] = vb.y;
            *(unsigned*)&t[4] = vb.z; *(unsigned*)&t[6] = vb.w;
            for (int e = 0; e < 8; ++e) sBt[(nc + e) * GLP + kr] = t[e];
        }
        if (kt + 1 < ksteps) {   // global_prefetch_b8 of next tiles
            __builtin_prefetch(A + (size_t)(mbase + (tid >> 1)) * K + (kt + 1) * 32, 0, 0);
            __builtin_prefetch(B + (size_t)((kt + 1) * 32 + (tid & 31)) * N + nbase, 0, 0);
        }
        stage_fence();
        __syncthreads();

        FragBF fa[2], fb[4];
        for (int mt = 0; mt < 2; ++mt) {
            const unsigned short* p = &sA[(wm + mt * 16 + lm) * GLP];
            for (int j = 0; j < 4; ++j) {
                fa[mt].u[j]     = *(const unsigned*)(p + kb + 2 * j);
                fa[mt].u[4 + j] = *(const unsigned*)(p + 16 + kb + 2 * j);
            }
        }
        for (int nt = 0; nt < 4; ++nt) {
            const unsigned short* p = &sBt[(wn + nt * 16 + lm) * GLP];
            for (int j = 0; j < 4; ++j) {
                fb[nt].u[j]     = *(const unsigned*)(p + kb + 2 * j);
                fb[nt].u[4 + j] = *(const unsigned*)(p + 16 + kb + 2 * j);
            }
        }
        for (int mt = 0; mt < 2; ++mt)
            for (int nt = 0; nt < 4; ++nt)
                acc[mt][nt] = __builtin_amdgcn_wmma_f32_16x16x32_bf16(
                    false, fa[mt].v, false, fb[nt].v, (short)0, acc[mt][nt], false, false);
        __syncthreads();
    }

    // --- epilogue ---
    for (int mt = 0; mt < 2; ++mt) {
        int rowb = mbase + wm + mt * 16 + (half ? 8 : 0);
        for (int nt = 0; nt < 4; ++nt) {
            int col = nbase + wn + nt * 16 + lm;
            float bv = 0.f;
            if (mode != 0 && col < biasN) bv = bias[col];
            for (int r = 0; r < 8; ++r) {
                float v = acc[mt][nt][r];
                if (mode != 0) v += bv;
                if (mode == 1) v = (v > 0.f) ? v : 0.2f * v;
                size_t off = (size_t)(rowb + r) * N + col;
                if (mode == 2) Cf[off] = v;
                else           Cb[off] = f2bf(v);
            }
        }
    }
}

// ---------------------------------------------------------------------------
// Flash attention: 64-query tile per block (4 waves x 16 rows), 32-key tiles,
// d = 256, online softmax in fp32, QK^T and P*V via bf16 WMMA.
// K tile staged by the Tensor Data Mover when available.
// ---------------------------------------------------------------------------
__global__ __launch_bounds__(128)
void flash_attn_wmma(const unsigned short* __restrict__ Qb,
                     const unsigned short* __restrict__ Kb,
                     const unsigned short* __restrict__ Vb,
                     float* __restrict__ Out,
                     int S, int causal, float scale)
{
    __shared__ __align__(16) unsigned short sK[32 * 264];   // K tile [key][d], 512B rows + 16B pad
    __shared__ __align__(16) unsigned short sVt[256 * 34];  // V tile [d][key]
    __shared__ __align__(16) float          sS[64 * 33];    // scores fp32
    __shared__ __align__(16) unsigned short sP[64 * 34];    // probs bf16
    __shared__ float sm[64], sl[64], sfac[64];

    const int tid = threadIdx.x, lane = tid & 31, wave = tid >> 5;
    const int half = lane >> 4, lm = lane & 15, kb = half * 8;
    const int b = blockIdx.y, qt = blockIdx.x;
    const size_t base = (size_t)b * S;

    // Q A-fragments for this wave's 16 rows, all 8 k-chunks of d=256
    FragBF aq[8];
    {
        const unsigned short* qp = Qb + (base + qt * 64 + wave * 16 + lm) * 256;
        for (int kc = 0; kc < 8; ++kc)
            for (int j = 0; j < 4; ++j) {
                aq[kc].u[j]     = *(const unsigned*)(qp + kc * 32 + kb + 2 * j);
                aq[kc].u[4 + j] = *(const unsigned*)(qp + kc * 32 + 16 + kb + 2 * j);
            }
    }
    v8f o[16];
    for (int i = 0; i < 16; ++i) o[i] = {};
    if (tid < 64) { sm[tid] = -3.0e38f; sl[tid] = 0.f; }
    __syncthreads();

    const int ktMax = causal ? ((qt * 64 + 63) >> 5) : (S / 32 - 1);
    for (int kt = 0; kt <= ktMax; ++kt) {
        const int kbeg = kt * 32;

#if HAVE_TDM
        // --- K tile via Tensor Data Mover: 2-D tile, data_size=8B units.
        // tile 64 units x 32 rows, row stride 64 units, LDS pad 4 DWORDs
        // every 128 DWORDs (512B row -> 528B pitch). D# layout per ISA ch.8.
        if (wave == 0) {
            typedef unsigned v4u __attribute__((ext_vector_type(4)));
            typedef int v8i __attribute__((ext_vector_type(8)));
            typedef int v4i __attribute__((ext_vector_type(4)));
            unsigned long long ga = (unsigned long long)(Kb + (base + kbeg) * 256);
            unsigned lds_off = (unsigned)(size_t)(__attribute__((address_space(3))) unsigned short*)sK;
            v4u g0 = { 1u,                                   // count=1, user D#
                       lds_off,
                       (unsigned)(ga & 0xFFFFFFFFu),
                       (unsigned)((ga >> 32) & 0x01FFFFFFu) | (2u << 30) }; // type=2
            v8i g1 = { (int)((3u << 16) | (1u << 20) | (6u << 22) | (3u << 25)),
                       (int)(64u << 16),          // tensor_dim0 = 64 units
                       (int)(16384u << 16),       // tensor_dim1 = 16384 rows
                       (int)(64u << 16),          // tile_dim0 = 64 units
                       32,                        // tile_dim1 = 32 rows
                       64, 0, 0 };                // tensor_dim0_stride = 64 units
            v4i gz = { 0, 0, 0, 0 };
# if __clang_major__ >= 23
            v8i gz8 = { 0, 0, 0, 0, 0, 0, 0, 0 };
            __builtin_amdgcn_tensor_load_to_lds(g0, g1, gz, gz, gz8, 0);
# else
            __builtin_amdgcn_tensor_load_to_lds(g0, g1, gz, gz, 0);
# endif
        }
#endif
        // --- stage V transposed (scatter) and, without TDM, K row-major ---
        for (int i = 0; i < 8; ++i) {
            int c = tid + 128 * i;           // 0..1023 chunks of 8 bf16
            int row = c >> 5;                // key 0..31
            int coff = (c & 31) * 8;         // d 0..248
#if !HAVE_TDM
            stage16(Kb + (base + kbeg + row) * 256 + coff, &sK[row * 264 + coff]);
#endif
            const uint4 vv = *(const uint4*)(Vb + (base + kbeg + row) * 256 + coff);
            unsigned short t[8];
            *(unsigned*)&t[0] = vv.x; *(unsigned*)&t[2] = vv.y;
            *(unsigned*)&t[4] = vv.z; *(unsigned*)&t[6] = vv.w;
            for (int e = 0; e < 8; ++e) sVt[(coff + e) * 34 + row] = t[e];
        }
        stage_fence();
#if HAVE_TDM
        if (wave == 0) __builtin_amdgcn_s_wait_tensorcnt(0);
#endif
        __syncthreads();

        // --- S = Q * K^T (16x32 strip per wave) ---
        v8f sacc[2]; sacc[0] = {}; sacc[1] = {};
        for (int kc = 0; kc < 8; ++kc) {
            for (int nt = 0; nt < 2; ++nt) {
                FragBF fb;
                const unsigned short* p = &sK[(nt * 16 + lm) * 264 + kc * 32];
                for (int j = 0; j < 4; ++j) {
                    fb.u[j]     = *(const unsigned*)(p + kb + 2 * j);
                    fb.u[4 + j] = *(const unsigned*)(p + 16 + kb + 2 * j);
                }
                sacc[nt] = __builtin_amdgcn_wmma_f32_16x16x32_bf16(
                    false, aq[kc].v, false, fb.v, (short)0, sacc[nt], false, false);
            }
        }
        {
            int rowb = wave * 16 + (half ? 8 : 0);
            for (int nt = 0; nt < 2; ++nt) {
                int col = nt * 16 + lm;
                for (int r = 0; r < 8; ++r)
                    sS[(rowb + r) * 33 + col] = sacc[nt][r] * scale;
            }
        }
        __syncthreads();

        // --- online softmax: 2 threads per row ---
        {
            int row = tid >> 1;
            int c0  = (tid & 1) * 16;
            int qg  = qt * 64 + row;
            float v[16], mx = -3.0e38f;
            for (int i = 0; i < 16; ++i) {
                float s = sS[row * 33 + c0 + i];
                if (causal && (kbeg + c0 + i > qg)) s = -__builtin_inff();
                v[i] = s; mx = fmaxf(mx, s);
            }
            mx = fmaxf(mx, __shfl_xor(mx, 1));
            float mNew = fmaxf(sm[row], mx);
            float sum = 0.f;
            for (int i = 0; i < 16; ++i) {
                float p = __expf(v[i] - mNew);
                sum += p;
                sP[row * 34 + c0 + i] = f2bf(p);
            }
            sum += __shfl_xor(sum, 1);
            if ((tid & 1) == 0) {
                float f = __expf(sm[row] - mNew);
                sfac[row] = f;
                sl[row] = sl[row] * f + sum;
                sm[row] = mNew;
            }
        }
        __syncthreads();

        // --- rescale O and accumulate P * V ---
        {
            float fac[8];
            int rowb = wave * 16 + (half ? 8 : 0);
            for (int r = 0; r < 8; ++r) fac[r] = sfac[rowb + r];
            for (int t = 0; t < 16; ++t)
                for (int r = 0; r < 8; ++r) o[t][r] *= fac[r];
            FragBF ap;
            const unsigned short* pp = &sP[(wave * 16 + lm) * 34];
            for (int j = 0; j < 4; ++j) {
                ap.u[j]     = *(const unsigned*)(pp + kb + 2 * j);
                ap.u[4 + j] = *(const unsigned*)(pp + 16 + kb + 2 * j);
            }
            for (int nt = 0; nt < 16; ++nt) {
                FragBF fv;
                const unsigned short* vp = &sVt[(nt * 16 + lm) * 34];
                for (int j = 0; j < 4; ++j) {
                    fv.u[j]     = *(const unsigned*)(vp + kb + 2 * j);
                    fv.u[4 + j] = *(const unsigned*)(vp + 16 + kb + 2 * j);
                }
                o[nt] = __builtin_amdgcn_wmma_f32_16x16x32_bf16(
                    false, ap.v, false, fv.v, (short)0, o[nt], false, false);
            }
        }
        __syncthreads();
    }

    // --- epilogue: O / l ---
    {
        int rowb = wave * 16 + (half ? 8 : 0);
        for (int r = 0; r < 8; ++r) {
            float linv = 1.f / sl[rowb + r];
            size_t gro = (base + qt * 64 + rowb + r) * 256;
            for (int nt = 0; nt < 16; ++nt)
                Out[gro + nt * 16 + lm] = o[nt][r] * linv;
        }
    }
}

// ---------------------------------------------------------------------------
// out = LayerNorm(a + res) * gamma + beta  (D = 256, eps = 1e-3)
// ---------------------------------------------------------------------------
__global__ __launch_bounds__(256)
void ln_residual(const float* __restrict__ a, const float* __restrict__ res,
                 const float* __restrict__ gamma, const float* __restrict__ beta,
                 float* __restrict__ outf, unsigned short* __restrict__ outb)
{
    __shared__ float red[256];
    const int row = blockIdx.x;
    const int i = threadIdx.x;
    const size_t off = (size_t)row * 256 + i;
    float v = a[off] + res[off];
    red[i] = v; __syncthreads();
    for (int s = 128; s > 0; s >>= 1) { if (i < s) red[i] += red[i + s]; __syncthreads(); }
    float mu = red[0] * (1.f / 256.f);
    __syncthreads();
    float d = v - mu;
    red[i] = d * d; __syncthreads();
    for (int s = 128; s > 0; s >>= 1) { if (i < s) red[i] += red[i + s]; __syncthreads(); }
    float var = red[0] * (1.f / 256.f);
    float o = d * rsqrtf(var + 1e-3f) * gamma[i] + beta[i];
    outf[off] = o;
    if (outb) outb[off] = f2bf(o);
}

// ---------------------------------------------------------------------------
extern "C" void kernel_launch(void* const* d_in, const int* in_sizes, int n_in,
                              void* d_out, int out_size, void* d_ws, size_t ws_size,
                              hipStream_t stream)
{
    (void)in_sizes; (void)n_in; (void)out_size; (void)ws_size;
    const int B = 4, S = 4096, D = 256, H = 400, Hp = 512;
    const size_t BS = (size_t)B * S;   // 16384 rows

    const float* inputs  = (const float*)d_in[0];
    const float* context = (const float*)d_in[1];
    const float* W[6] = { (const float*)d_in[2], (const float*)d_in[3], (const float*)d_in[4],
                          (const float*)d_in[5], (const float*)d_in[6], (const float*)d_in[7] };
    const float* W1 = (const float*)d_in[8];
    const float* b1 = (const float*)d_in[9];
    const float* W2 = (const float*)d_in[10];
    const float* b2 = (const float*)d_in[11];
    const float* gamma = (const float*)d_in[12];
    const float* beta  = (const float*)d_in[13];
    float* out = (float*)d_out;

    // workspace carve-out (~144 MB)
    char* ws = (char*)d_ws;
    size_t off = 0;
    auto alloc = [&](size_t bytes) -> char* {
        char* p = ws + off; off += (bytes + 255) & ~(size_t)255; return p;
    };
    unsigned short* inb  = (unsigned short*)alloc(BS * D * 2);
    unsigned short* ctxb = (unsigned short*)alloc(BS * D * 2);
    unsigned short* wb[6];
    for (int i = 0; i < 6; ++i) wb[i] = (unsigned short*)alloc((size_t)D * D * 2);
    unsigned short* w1b = (unsigned short*)alloc((size_t)D * Hp * 2);
    unsigned short* w2b = (unsigned short*)alloc((size_t)Hp * D * 2);
    unsigned short* Kb  = (unsigned short*)alloc(BS * D * 2);
    unsigned short* Vb  = (unsigned short*)alloc(BS * D * 2);
    unsigned short* Qb  = (unsigned short*)alloc(BS * D * 2);
    float*          attn= (float*)alloc(BS * D * 4);
    float*          xf  = (float*)alloc(BS * D * 4);
    unsigned short* xb  = (unsigned short*)alloc(BS * D * 2);
    float*          yf  = (float*)alloc(BS * D * 4);
    unsigned short* yb  = (unsigned short*)alloc(BS * D * 2);
    unsigned short* hb  = (unsigned short*)alloc(BS * Hp * 2);
    float*          ff  = (float*)alloc(BS * D * 4);

    auto cast = [&](const float* s, unsigned short* dptr, int sr, int sc, int dr, int dc) {
        int tot = dr * dc;
        cast_pad_bf16<<<(tot + 255) / 256, 256, 0, stream>>>(s, dptr, sr, sc, dr, dc);
    };
    cast(inputs,  inb,  1, (int)(BS * D), 1, (int)(BS * D));
    cast(context, ctxb, 1, (int)(BS * D), 1, (int)(BS * D));
    for (int i = 0; i < 6; ++i) cast(W[i], wb[i], 1, D * D, 1, D * D);
    cast(W1, w1b, D, H, D, Hp);     // zero-pad columns 400..511
    cast(W2, w2b, H, D, Hp, D);     // zero-pad rows    400..511

    dim3 gproj(D / 128, (int)(BS / 128));   // (2, 128)
    dim3 gattn(S / 64, B);                  // (64, 4)
    const float scale = 1.0f / 64.0f;       // 1/sqrt(4096)

    // ---- 1) causal self-attention ----
    gemm_bf16_wmma<<<gproj, 256, 0, stream>>>(inb, wb[0], nullptr, 0, nullptr, Kb, (int)BS, D, D, 0);
    gemm_bf16_wmma<<<gproj, 256, 0, stream>>>(inb, wb[1], nullptr, 0, nullptr, Vb, (int)BS, D, D, 0);
    gemm_bf16_wmma<<<gproj, 256, 0, stream>>>(inb, wb[2], nullptr, 0, nullptr, Qb, (int)BS, D, D, 0);
    flash_attn_wmma<<<gattn, 128, 0, stream>>>(Qb, Kb, Vb, attn, S, 1, scale);
    ln_residual<<<(int)BS, 256, 0, stream>>>(attn, inputs, gamma, beta, xf, xb);

    // ---- 2) cross-attention (queries = x, keys/values = context) ----
    gemm_bf16_wmma<<<gproj, 256, 0, stream>>>(ctxb, wb[3], nullptr, 0, nullptr, Kb, (int)BS, D, D, 0);
    gemm_bf16_wmma<<<gproj, 256, 0, stream>>>(ctxb, wb[4], nullptr, 0, nullptr, Vb, (int)BS, D, D, 0);
    gemm_bf16_wmma<<<gproj, 256, 0, stream>>>(xb,   wb[5], nullptr, 0, nullptr, Qb, (int)BS, D, D, 0);
    flash_attn_wmma<<<gattn, 128, 0, stream>>>(Qb, Kb, Vb, attn, S, 0, scale);
    ln_residual<<<(int)BS, 256, 0, stream>>>(attn, xf, gamma, beta, yf, yb);

    // ---- 3) FFN ----
    dim3 g1(Hp / 128, (int)(BS / 128));     // (4, 128)
    gemm_bf16_wmma<<<g1, 256, 0, stream>>>(yb, w1b, b1, H, nullptr, hb, (int)BS, Hp, D, 1);
    dim3 g2(D / 128, (int)(BS / 128));      // (2, 128)
    gemm_bf16_wmma<<<g2, 256, 0, stream>>>(hb, w2b, b2, D, ff, nullptr, (int)BS, D, Hp, 2);
    ln_residual<<<(int)BS, 256, 0, stream>>>(ff, yf, gamma, beta, out, nullptr);
}
